// MHSA_block_26663156973895
// MI455X (gfx1250) — matrix operand order repaired and checked by reading, not compiled, not run on hardware
//
#include <hip/hip_runtime.h>
#include <math.h>

typedef __attribute__((ext_vector_type(16))) __bf16 v16bf;
typedef __attribute__((ext_vector_type(8)))  float  v8f;
typedef __attribute__((ext_vector_type(4)))  unsigned int u32x4;
typedef __attribute__((ext_vector_type(8)))  int   i32x8;
typedef __attribute__((ext_vector_type(4)))  int   i32x4;
typedef unsigned short u16;
typedef unsigned int   u32;

#define H_   8
#define DF_  64
#define P_   2048
#define B_   4
#define D_   512
#define PT   32           // query rows per block
#define QT   32           // key cols per iteration
#define LOG2E 1.4426950408889634f

// Tensor Data Mover available? (device pass only; host pass uses fallback copies)
#if defined(__HIP_DEVICE_COMPILE__) && __has_builtin(__builtin_amdgcn_tensor_load_to_lds) && \
    __has_builtin(__builtin_amdgcn_s_wait_tensorcnt)
#define USE_TDM 1
#else
#define USE_TDM 0
#endif

union Frag16 { v16bf v; u16 u[16]; uint4 q4[2]; };
union Pack8  { u16 u[8]; uint4 q; };

__device__ __forceinline__ u16 f2bf(float f) {
  u32 x = __float_as_uint(f);
  x += 0x7FFFu + ((x >> 16) & 1u);       // round-to-nearest-even
  return (u16)(x >> 16);
}

__device__ __forceinline__ v8f wmma_bf16(v16bf a, v16bf b, v8f c) {
  // D = A(16x32 bf16) * B(32x16 bf16) + C(16x16 f32)
  return __builtin_amdgcn_wmma_f32_16x16x32_bf16(false, a, false, b, (short)0, c,
                                                 false, false);
}

#if USE_TDM
// 2D tile load, global -> LDS, bf16 elements (data_size=2B), with LDS row padding.
//   lds_off     : byte offset in LDS segment (dynamic-LDS base assumed 0)
//   gsrc        : global address of tile start
//   tw, th      : tile width (elements) / height (rows)
//   stride      : row stride in elements
//   pint, pamt  : pad_interval / pad_amount codes (ISA D# group1 encodings)
__device__ __forceinline__ void tdm_load_2d(unsigned lds_off, const void* gsrc,
                                            unsigned tw, unsigned th,
                                            unsigned stride, unsigned pint,
                                            unsigned pamt) {
  unsigned long long ga = (unsigned long long)(size_t)gsrc;
  u32x4 g0;
  g0[0] = 1u;                                          // count=1, user mode
  g0[1] = lds_off;                                     // lds_addr
  g0[2] = (unsigned)(ga & 0xFFFFFFFFu);                // global_addr lo
  g0[3] = (unsigned)((ga >> 32) & 0x01FFFFFFu) | (2u << 30);  // addr hi | type=2
  i32x8 g1;
  g1[0] = (int)((1u << 16) | (1u << 20) | (pint << 22) | (pamt << 25));
  g1[1] = (int)((tw & 0xFFFFu) << 16);                 // tensor_dim0 lo16
  g1[2] = (int)((tw >> 16) | ((th & 0xFFFFu) << 16));  // dim0 hi | tensor_dim1 lo
  g1[3] = (int)((th >> 16) | (tw << 16));              // dim1 hi | tile_dim0
  g1[4] = (int)th;                                     // tile_dim1 (tile_dim2=0)
  g1[5] = (int)stride;                                 // tensor_dim0_stride lo32
  g1[6] = 0;
  g1[7] = 0;
  i32x4 gz; gz[0] = 0; gz[1] = 0; gz[2] = 0; gz[3] = 0;
#if __clang_major__ >= 23
  i32x8 gz8; gz8[0]=0; gz8[1]=0; gz8[2]=0; gz8[3]=0; gz8[4]=0; gz8[5]=0; gz8[6]=0; gz8[7]=0;
  __builtin_amdgcn_tensor_load_to_lds(g0, g1, gz, gz, gz8, 0);
#else
  __builtin_amdgcn_tensor_load_to_lds(g0, g1, gz, gz, 0);
#endif
}
#endif

// ---------------------------------------------------------------------------
// Generic WMMA GEMM:  C[M,N] = scale * (A[M,K] @ Bw[K,N])
//   AF32=1: A is f32 (converted to bf16 while staging); AF32=0: A is bf16
//   MODE 0: write bf16 head-major workspace layout [b][h][p][df]
//   MODE 1: write f32 row-major [r][c]
// Block: 256 threads (8 waves as 2x4), tile 64(M) x 64(N), K step 32.
// Dynamic LDS: Als @0 (64x40 u16), Bls @5120 (64x40 u16) -> 10240 B.
// ---------------------------------------------------------------------------
template <int AF32, int MODE>
__global__ __launch_bounds__(256)
void gemm_kernel(const void* __restrict__ Aptr, const float* __restrict__ Bw,
                 float scale, void* __restrict__ outp, int M, int N, int K) {
  extern __shared__ __align__(16) char gsm[];
  u16* Als = (u16*)gsm;              // [m][k], row stride 40
  u16* Bls = (u16*)(gsm + 5120);     // [n][k] (transposed), stride 40

  const int t    = threadIdx.x;
  const int wave = t >> 5;
  const int lane = t & 31;
  const int half = lane >> 4;
  const int l16  = lane & 15;
  const int wm   = wave >> 2;    // 0..1
  const int wn   = wave & 3;     // 0..3
  const int m0   = blockIdx.y * 64;
  const int n0   = blockIdx.x * 64;

  v8f acc[2] = { {0,0,0,0,0,0,0,0}, {0,0,0,0,0,0,0,0} };

  for (int k0 = 0; k0 < K; k0 += 32) {
    // --- stage A tile 64x32 -> bf16 LDS ---
    if constexpr (AF32) {
      int r  = t >> 2;
      int c8 = (t & 3) << 3;
      const float* a = (const float*)Aptr + (size_t)(m0 + r) * K + k0 + c8;
      Pack8 pk;
      #pragma unroll
      for (int i = 0; i < 8; ++i) pk.u[i] = f2bf(a[i]);
      *(uint4*)&Als[r * 40 + c8] = pk.q;
    } else {
#if USE_TDM
      if (wave == 0)   // one TDM per block: 64 rows x 32 bf16, pad 64B->80B rows
        tdm_load_2d(0u, (const u16*)Aptr + (size_t)m0 * K + k0,
                    32u, 64u, (unsigned)K, 3u, 3u);
#else
      int r  = t >> 2;
      int c8 = (t & 3) << 3;
      const u16* a = (const u16*)Aptr + (size_t)(m0 + r) * K + k0 + c8;
      *(uint4*)&Als[r * 40 + c8] = *(const uint4*)a;
#endif
    }
    // --- stage B tile 32x64 transposed -> [n][k] ---
    {
      int kk  = t & 31;
      int nn0 = (t >> 5) << 3;
      const float* bsrc = Bw + (size_t)(k0 + kk) * N + n0 + nn0;
      #pragma unroll
      for (int i = 0; i < 8; ++i) Bls[(nn0 + i) * 40 + kk] = f2bf(bsrc[i]);
    }
#if USE_TDM
    if constexpr (!AF32) {
      if (wave == 0) __builtin_amdgcn_s_wait_tensorcnt(0);
    }
#endif
    __syncthreads();

    // --- fragments + WMMA ---
    Frag16 bf;
    {
      const u16* bb = &Bls[(wn * 16 + l16) * 40 + half * 8]; // k = half*8 + e, 16 + half*8 + e
      bf.q4[0] = *(const uint4*)bb;
      bf.q4[1] = *(const uint4*)(bb + 16);
    }
    #pragma unroll
    for (int tM = 0; tM < 2; ++tM) {
      Frag16 af;
      const u16* ab = &Als[(wm * 32 + tM * 16 + l16) * 40 + half * 8];
      af.q4[0] = *(const uint4*)ab;          // k = half*8 + e      (e=0..7)
      af.q4[1] = *(const uint4*)(ab + 16);   // k = 16 + half*8 + e (e=8..15)
      acc[tM] = wmma_bf16(af.v, bf.v, acc[tM]);
    }
    __syncthreads();
  }

  // --- epilogue ---
  #pragma unroll
  for (int tM = 0; tM < 2; ++tM) {
    #pragma unroll
    for (int e = 0; e < 8; ++e) {
      int r = m0 + wm * 32 + tM * 16 + half * 8 + e;
      int c = n0 + wn * 16 + l16;
      float v = acc[tM][e] * scale;
      if constexpr (MODE == 0) {
        int b = r >> 11, p = r & 2047, hh = c >> 6, df = c & 63;
        ((u16*)outp)[(((size_t)b * H_ + hh) * P_ + p) * DF_ + df] = f2bf(v);
      } else {
        ((float*)outp)[(size_t)r * N + c] = v;
      }
    }
  }
}

// ---------------------------------------------------------------------------
// Flash-style talking-heads attention.
// grid = B * (P/PT) blocks, 256 threads = 8 waves, wave h owns head h.
// LDS:
//   reg0: 8 slabs x 4608B  (Q stage / K stage / S_raw f32 / P bf16, reused)
//   vreg: 8 slabs x 5120B  (V transposed [df][q])
// Q/K tiles staged by the Tensor Data Mover (32 rows x 128B, +16B LDS row pad).
// ---------------------------------------------------------------------------
__global__ __launch_bounds__(256)
void attn_kernel(const u16* __restrict__ qws, const u16* __restrict__ kws,
                 const u16* __restrict__ vws, const float* __restrict__ bias,
                 const float* __restrict__ wtalk, u16* __restrict__ ows) {
  extern __shared__ __align__(16) char smem[];
  u16* reg0 = (u16*)smem;                       // 8 * 4608 B = 36864 B
  u16* vreg = (u16*)(smem + 36864);             // 8 * 5120 B = 40960 B

  const int t    = threadIdx.x;
  const int h    = t >> 5;                      // wave == head
  const int lane = t & 31;
  const int half = lane >> 4;
  const int l16  = lane & 15;
  const int blk  = blockIdx.x;
  const int b    = blk / (P_ / PT);
  const int p0   = (blk % (P_ / PT)) * PT;
#if USE_TDM
  const int hu   = __builtin_amdgcn_readfirstlane(h);  // wave-uniform head id
#endif

  // Wtalk row for this (output) head g = h:  mixed_g = sum_h wt[h] * S_h
  float wt[H_];
  #pragma unroll
  for (int i = 0; i < H_; ++i) wt[i] = wtalk[h * H_ + i];

  // ---- stage Q tile (wave-private slabs), load A-fragments ----
#if USE_TDM
  tdm_load_2d((unsigned)(hu * 4608),
              qws + ((size_t)(b * H_ + hu) * P_ + p0) * DF_,
              64u, 32u, 64u, 4u, 3u);
  __builtin_amdgcn_s_wait_tensorcnt(0);
#else
  {
    int h2 = t >> 5, pr = t & 31;
    const u16* src = qws + ((size_t)(b * H_ + h2) * P_ + (p0 + pr)) * DF_;
    uint4* d4 = (uint4*)(reg0 + h2 * 2304 + pr * 72);
    const uint4* s4 = (const uint4*)src;
    #pragma unroll
    for (int i = 0; i < 8; ++i) d4[i] = s4[i];
  }
#endif
  Frag16 qf[2][2];
  #pragma unroll
  for (int tM = 0; tM < 2; ++tM)
    #pragma unroll
    for (int tK = 0; tK < 2; ++tK) {
      const u16* base = reg0 + h * 2304 + (tM * 16 + l16) * 72 + tK * 32 + half * 8;
      qf[tM][tK].q4[0] = *(const uint4*)base;
      qf[tM][tK].q4[1] = *(const uint4*)(base + 16);
    }
#if USE_TDM
  // Drain Q fragment ds_loads before the K TDM reuses this slab.
  asm volatile("s_wait_dscnt 0x0" ::: "memory");
#endif

  // ---- running state ----
  v8f Oacc[2][4];
  #pragma unroll
  for (int i = 0; i < 2; ++i)
    #pragma unroll
    for (int j = 0; j < 4; ++j) Oacc[i][j] = (v8f){0,0,0,0,0,0,0,0};
  float mrow[2][8], lrow[2][8];
  #pragma unroll
  for (int i = 0; i < 2; ++i)
    #pragma unroll
    for (int e = 0; e < 8; ++e) { mrow[i][e] = -INFINITY; lrow[i][e] = 0.f; }

  for (int q0 = 0; q0 < P_; q0 += QT) {
    // ---- prefetch next tiles (global_prefetch_b8) ----
    if (q0 + QT < P_) {
      __builtin_prefetch(kws + ((size_t)(b * H_ + h) * P_ + (q0 + QT + lane)) * DF_, 0, 0);
      __builtin_prefetch(vws + ((size_t)(b * H_ + h) * P_ + (q0 + QT + lane)) * DF_, 0, 0);
      __builtin_prefetch(bias + ((size_t)h * P_ + (p0 + lane)) * P_ + q0 + QT, 0, 0);
    }

    // ---- stage K via TDM (wave-private slab), V transposed by threads ----
#if USE_TDM
    tdm_load_2d((unsigned)(hu * 4608),
                kws + ((size_t)(b * H_ + hu) * P_ + q0) * DF_,
                64u, 32u, 64u, 4u, 3u);
#else
    {
      int h2 = t >> 5, qr = t & 31;
      const u16* ks = kws + ((size_t)(b * H_ + h2) * P_ + (q0 + qr)) * DF_;
      uint4* kd = (uint4*)(reg0 + h2 * 2304 + qr * 72);
      const uint4* ks4 = (const uint4*)ks;
      #pragma unroll
      for (int i = 0; i < 8; ++i) kd[i] = ks4[i];
    }
#endif
    {
      int h2 = t >> 5, qr = t & 31;
      const u16* vs = vws + ((size_t)(b * H_ + h2) * P_ + (q0 + qr)) * DF_;
      u16* vbase = vreg + h2 * 2560;            // [df][q], stride 40
      const uint4* vs4 = (const uint4*)vs;
      #pragma unroll
      for (int i = 0; i < 8; ++i) {
        Pack8 pk; pk.q = vs4[i];
        #pragma unroll
        for (int j = 0; j < 8; ++j) vbase[(i * 8 + j) * 40 + qr] = pk.u[j];
      }
    }
#if USE_TDM
    __builtin_amdgcn_s_wait_tensorcnt(0);
#endif

    // ---- S = Q @ K^T  (per head) ----
    v8f S[2][2];
    #pragma unroll
    for (int tN = 0; tN < 2; ++tN) {
      Frag16 kf[2];
      #pragma unroll
      for (int tK = 0; tK < 2; ++tK) {
        const u16* base = reg0 + h * 2304 + (tN * 16 + l16) * 72 + tK * 32 + half * 8;
        kf[tK].q4[0] = *(const uint4*)base;
        kf[tK].q4[1] = *(const uint4*)(base + 16);
      }
      #pragma unroll
      for (int tM = 0; tM < 2; ++tM) {
        v8f a = (v8f){0,0,0,0,0,0,0,0};
        a = wmma_bf16(qf[tM][0].v, kf[0].v, a);
        a = wmma_bf16(qf[tM][1].v, kf[1].v, a);
        S[tM][tN] = a;
      }
    }

    // ---- publish raw S (f32) for cross-head mixing ----
    float* sbase = (float*)reg0 + h * 1152;     // [32][36]
    #pragma unroll
    for (int tM = 0; tM < 2; ++tM)
      #pragma unroll
      for (int tN = 0; tN < 2; ++tN)
        #pragma unroll
        for (int e = 0; e < 8; ++e)
          sbase[(tM * 16 + half * 8 + e) * 36 + tN * 16 + l16] = S[tM][tN][e];
    __syncthreads();

    // ---- talking-heads mix + bias ----
    float mixed[2][2][8];
    #pragma unroll
    for (int tM = 0; tM < 2; ++tM)
      #pragma unroll
      for (int tN = 0; tN < 2; ++tN)
        #pragma unroll
        for (int e = 0; e < 8; ++e) {
          int row = tM * 16 + half * 8 + e;
          int col = tN * 16 + l16;
          float a = 0.f;
          #pragma unroll
          for (int hh = 0; hh < H_; ++hh)
            a += wt[hh] * ((float*)reg0)[hh * 1152 + row * 36 + col];
          a += bias[((size_t)h * P_ + (p0 + row)) * P_ + (q0 + col)];
          mixed[tM][tN][e] = a;
        }
    __syncthreads();   // all waves done reading S_raw; slabs now reusable

    // ---- online softmax update ----
    float sc_arr[2][8];
    #pragma unroll
    for (int tM = 0; tM < 2; ++tM)
      #pragma unroll
      for (int e = 0; e < 8; ++e) {
        float v = fmaxf(mixed[tM][0][e], mixed[tM][1][e]);
        v = fmaxf(v, __shfl_xor(v, 1));
        v = fmaxf(v, __shfl_xor(v, 2));
        v = fmaxf(v, __shfl_xor(v, 4));
        v = fmaxf(v, __shfl_xor(v, 8));
        float mo = mrow[tM][e];
        float mn = fmaxf(mo, v);
        float sc = exp2f((mo - mn) * LOG2E);
        float rs = 0.f;
        #pragma unroll
        for (int tN = 0; tN < 2; ++tN) {
          float pv = exp2f((mixed[tM][tN][e] - mn) * LOG2E);
          mixed[tM][tN][e] = pv;
          rs += pv;
        }
        rs += __shfl_xor(rs, 1);
        rs += __shfl_xor(rs, 2);
        rs += __shfl_xor(rs, 4);
        rs += __shfl_xor(rs, 8);
        lrow[tM][e] = lrow[tM][e] * sc + rs;
        mrow[tM][e] = mn;
        sc_arr[tM][e] = sc;
      }
    #pragma unroll
    for (int tM = 0; tM < 2; ++tM)
      #pragma unroll
      for (int tNv = 0; tNv < 4; ++tNv)
        #pragma unroll
        for (int e = 0; e < 8; ++e)
          Oacc[tM][tNv][e] *= sc_arr[tM][e];

    // ---- P (bf16) via own slab, O += P @ V ----
    u16* pbase = reg0 + h * 2304;               // [32][40]
    #pragma unroll
    for (int tM = 0; tM < 2; ++tM)
      #pragma unroll
      for (int tN = 0; tN < 2; ++tN)
        #pragma unroll
        for (int e = 0; e < 8; ++e)
          pbase[(tM * 16 + half * 8 + e) * 40 + tN * 16 + l16] = f2bf(mixed[tM][tN][e]);

    Frag16 pf[2];
    #pragma unroll
    for (int tM = 0; tM < 2; ++tM) {
      const u16* base = pbase + (tM * 16 + l16) * 40 + half * 8;
      pf[tM].q4[0] = *(const uint4*)base;
      pf[tM].q4[1] = *(const uint4*)(base + 16);
    }
    #pragma unroll
    for (int tNv = 0; tNv < 4; ++tNv) {
      Frag16 vf;
      const u16* vb = vreg + h * 2560 + (tNv * 16 + l16) * 40 + half * 8;
      vf.q4[0] = *(const uint4*)vb;
      vf.q4[1] = *(const uint4*)(vb + 16);
      #pragma unroll
      for (int tM = 0; tM < 2; ++tM)
        Oacc[tM][tNv] = wmma_bf16(pf[tM].v, vf.v, Oacc[tM][tNv]);
    }
  }

  // ---- epilogue: normalize, write bf16 [b][p][h*64+df] ----
  #pragma unroll
  for (int tM = 0; tM < 2; ++tM)
    #pragma unroll
    for (int tNv = 0; tNv < 4; ++tNv)
      #pragma unroll
      for (int e = 0; e < 8; ++e) {
        int row = p0 + tM * 16 + half * 8 + e;
        int col = h * DF_ + tNv * 16 + l16;
        float v = Oacc[tM][tNv][e] / lrow[tM][e];
        ows[((size_t)b * P_ + row) * (H_ * DF_) + col] = f2bf(v);
      }
}

// ---------------------------------------------------------------------------
extern "C" void kernel_launch(void* const* d_in, const int* in_sizes, int n_in,
                              void* d_out, int out_size, void* d_ws, size_t ws_size,
                              hipStream_t stream) {
  (void)in_sizes; (void)n_in; (void)out_size; (void)ws_size;
  const float* x     = (const float*)d_in[0];
  const float* bias  = (const float*)d_in[1];
  const float* Wq    = (const float*)d_in[2];
  const float* Wk    = (const float*)d_in[3];
  const float* Wv    = (const float*)d_in[4];
  const float* Wtalk = (const float*)d_in[5];
  const float* Wo    = (const float*)d_in[6];

  const size_t elems = (size_t)B_ * H_ * P_ * DF_;   // 4,194,304 bf16 per tensor
  u16* qws = (u16*)d_ws;
  u16* kws = qws + elems;
  u16* vws = kws + elems;
  u16* ows = vws + elems;                            // [b][p][512] bf16

  const int M = B_ * P_, N = H_ * DF_, K = D_;
  dim3 gblk(N / 64, M / 64);                         // (8, 128)
  const size_t glds = 10240;                         // Als + Bls

  // Q/K/V projections (Q scaled by 1/sqrt(DF) = 0.125)
  gemm_kernel<1, 0><<<gblk, 256, glds, stream>>>(x, Wq, 0.125f, qws, M, N, K);
  gemm_kernel<1, 0><<<gblk, 256, glds, stream>>>(x, Wk, 1.0f,   kws, M, N, K);
  gemm_kernel<1, 0><<<gblk, 256, glds, stream>>>(x, Wv, 1.0f,   vws, M, N, K);

  // Flash talking-heads attention
  const size_t lds = 36864 + 40960;                  // 77824 B dynamic LDS
  attn_kernel<<<B_ * (P_ / PT), 256, lds, stream>>>(qws, kws, vws, bias, Wtalk, ows);

  // Output projection: out = O @ Wo  (f32 out)
  gemm_kernel<0, 1><<<gblk, 256, glds, stream>>>(ows, Wo, 1.0f, d_out, M, D_, N);
}
